// IONMGNNModel_41575283425427
// MI455X (gfx1250) — matrix operand order repaired and checked by reading, not compiled, not run on hardware
//
#include <hip/hip_runtime.h>
#include <math.h>

// ---------------- problem constants ----------------
#define N_NODES 100000
#define N_EDGES 1600000
#define D_IN    64
#define H1_DIM  128
#define H2_DIM  64
#define ATT_DIM 32

// ---------------- types ----------------
typedef __attribute__((ext_vector_type(16))) __bf16 v16bf;
typedef __attribute__((ext_vector_type(8)))  float  v8f;

// ---------------- helpers ----------------
__device__ __forceinline__ void atomAddF(float* p, float v) {
  __hip_atomic_fetch_add(p, v, __ATOMIC_RELAXED, __HIP_MEMORY_SCOPE_AGENT);
}
__device__ __forceinline__ void atomAddLds(float* p, float v) {
  __hip_atomic_fetch_add(p, v, __ATOMIC_RELAXED, __HIP_MEMORY_SCOPE_WORKGROUP);
}

// A fragment: 16x32 bf16 of row-major [M x K] source, ld = lda.
// Lane L: m = m0 + (L&15), kbase = (L>>4)*8.
// halves 0..7  -> k = ko + kbase + i        (contiguous, 16B)
// halves 8..15 -> k = ko + 16 + kbase + i-8 (contiguous, 16B)
__device__ __forceinline__ v16bf load_a_frag(const __bf16* A, int lda, int m0,
                                             int ko, int lane) {
  const int m = m0 + (lane & 15);
  const int kbase = (lane >> 4) << 3;
  const __bf16* p = A + (size_t)m * lda + ko + kbase;
  v16bf a;
#pragma unroll
  for (int i = 0; i < 8; ++i) a[i] = p[i];
#pragma unroll
  for (int i = 0; i < 8; ++i) a[8 + i] = p[16 + i];
  return a;
}

// B fragment from TRANSPOSED weights Bt[Nout x K] (bf16), ld = ldk.
// Lane L: n = n0 + (L&15), kgroup = (L>>4)*8; halves 0..7 -> k = ko + kgroup + i,
// halves 8..15 -> k = ko + 16 + kgroup + i-8 (two 16B loads).
__device__ __forceinline__ v16bf load_bt_frag(const __bf16* Bt, int ldk, int n0,
                                              int ko, int lane) {
  const int n = n0 + (lane & 15);
  const int kg = (lane >> 4) << 3;
  const __bf16* p = Bt + (size_t)n * ldk + ko + kg;
  v16bf b;
#pragma unroll
  for (int i = 0; i < 8; ++i) b[i] = p[i];
#pragma unroll
  for (int i = 0; i < 8; ++i) b[8 + i] = p[16 + i];
  return b;
}

#define WMMA_BF16(A, B, C)                                                    \
  __builtin_amdgcn_wmma_f32_16x16x32_bf16(false, (A), false, (B), (short)0,   \
                                          (C), false, false)

// ---------------- elementwise / graph kernels ----------------
__global__ void cvt_bf16_kernel(const float* __restrict__ src,
                                __bf16* __restrict__ dst, int n) {
  int i = blockIdx.x * blockDim.x + threadIdx.x;
  if (i < n) dst[i] = (__bf16)src[i];
}

// dst[Nout x K] (bf16) = transpose of src[K x Nout] (f32). logK: K = 1<<logK.
__global__ void cvt_transpose_kernel(const float* __restrict__ src,
                                     __bf16* __restrict__ dst, int logK,
                                     int nout, int total) {
  int i = blockIdx.x * blockDim.x + threadIdx.x;
  if (i >= total) return;
  int n = i >> logK;
  int k = i & ((1 << logK) - 1);
  dst[i] = (__bf16)src[(size_t)k * nout + n];
}

__global__ void deg_kernel(const int* __restrict__ dst, float* __restrict__ deg,
                           int e) {
  int i = blockIdx.x * blockDim.x + threadIdx.x;
  if (i < e) atomAddF(&deg[dst[i]], 1.0f);
}

// One thread handles 4 consecutive features of an edge: one index-pair load,
// one vector feature load (b128 f32 / b64 bf16), 4 native f32 atomics.
template <typename T>
__global__ void scatter_add_kernel(const T* __restrict__ feat,
                                   const int* __restrict__ src,
                                   const int* __restrict__ dst,
                                   float* __restrict__ out, int e, int logF) {
  const int F = 1 << logF;
  const int chunks = F >> 2;                     // groups of 4 features
  long long idx = (long long)blockIdx.x * blockDim.x + threadIdx.x;
  if (idx >= (long long)e * chunks) return;
  int ed = (int)(idx >> (logF - 2));
  int f4 = ((int)idx & (chunks - 1)) << 2;
  int s = src[ed], d = dst[ed];
  const T* fp = feat + (size_t)s * F + f4;
  float v0 = (float)fp[0], v1 = (float)fp[1], v2 = (float)fp[2],
        v3 = (float)fp[3];
  float* op = out + (size_t)d * F + f4;
  atomAddF(op + 0, v0);
  atomAddF(op + 1, v1);
  atomAddF(op + 2, v2);
  atomAddF(op + 3, v3);
}

__global__ void finalize_kernel(const float* __restrict__ sum,
                                const float* __restrict__ deg,
                                __bf16* __restrict__ out, long long total,
                                int logF) {
  long long i = (long long)blockIdx.x * blockDim.x + threadIdx.x;
  if (i >= total) return;
  int n = (int)(i >> logF);
  out[i] = (__bf16)(sum[i] / fmaxf(deg[n], 1.0f));
}

// ---------------- GEMM layer 1: h1 = relu(agg1@W1_l + x@W1_r + b1) ----------
__global__ void gemm1_kernel(const __bf16* __restrict__ agg,
                             const __bf16* __restrict__ x,
                             const __bf16* __restrict__ Wlt,
                             const __bf16* __restrict__ Wrt,
                             const float* __restrict__ b1,
                             __bf16* __restrict__ h1) {
  const int wave = threadIdx.x >> 5, lane = threadIdx.x & 31;
  const int tile = blockIdx.x * 8 + wave;       // (N/16)*(H1/16) = 6250*8
  const int m0 = (tile >> 3) << 4;
  const int n0 = (tile & 7) << 4;
  v8f c = {};
#pragma unroll
  for (int ko = 0; ko < D_IN; ko += 32) {
    v16bf a = load_a_frag(agg, D_IN, m0, ko, lane);
    v16bf b = load_bt_frag(Wlt, D_IN, n0, ko, lane);
    c = WMMA_BF16(a, b, c);
    a = load_a_frag(x, D_IN, m0, ko, lane);
    b = load_bt_frag(Wrt, D_IN, n0, ko, lane);
    c = WMMA_BF16(a, b, c);
  }
  const int col = n0 + (lane & 15);
  const float bias = b1[col];
#pragma unroll
  for (int r = 0; r < 8; ++r) {
    int row = m0 + ((lane >> 4) << 3) + r;
    float v = fmaxf(c[r] + bias, 0.0f);
    h1[(size_t)row * H1_DIM + col] = (__bf16)v;
  }
}

// ---------------- GEMM layer 2: h2 = relu(agg2@W2_l + h1@W2_r + b2) ---------
__global__ void gemm2_kernel(const __bf16* __restrict__ agg,
                             const __bf16* __restrict__ h1,
                             const __bf16* __restrict__ Wlt,
                             const __bf16* __restrict__ Wrt,
                             const float* __restrict__ b2,
                             float* __restrict__ h2f,
                             __bf16* __restrict__ h2b) {
  const int wave = threadIdx.x >> 5, lane = threadIdx.x & 31;
  const int tile = blockIdx.x * 8 + wave;       // 6250*4 tiles
  const int m0 = (tile >> 2) << 4;
  const int n0 = (tile & 3) << 4;
  v8f c = {};
#pragma unroll
  for (int ko = 0; ko < H1_DIM; ko += 32) {
    v16bf a = load_a_frag(agg, H1_DIM, m0, ko, lane);
    v16bf b = load_bt_frag(Wlt, H1_DIM, n0, ko, lane);
    c = WMMA_BF16(a, b, c);
    a = load_a_frag(h1, H1_DIM, m0, ko, lane);
    b = load_bt_frag(Wrt, H1_DIM, n0, ko, lane);
    c = WMMA_BF16(a, b, c);
  }
  const int col = n0 + (lane & 15);
  const float bias = b2[col];
#pragma unroll
  for (int r = 0; r < 8; ++r) {
    int row = m0 + ((lane >> 4) << 3) + r;
    float v = fmaxf(c[r] + bias, 0.0f);
    h2f[(size_t)row * H2_DIM + col] = v;
    h2b[(size_t)row * H2_DIM + col] = (__bf16)v;
  }
}

// ------- attention: w_raw = tanh(h2@Wa+ba)@ctx + 0.4*x[:,63]  ---------------
__global__ void gemm3_kernel(const __bf16* __restrict__ h2,
                             const __bf16* __restrict__ Wat,
                             const float* __restrict__ ba,
                             const float* __restrict__ ctx,
                             const float* __restrict__ x,
                             float* __restrict__ w_raw) {
  const int wave = threadIdx.x >> 5, lane = threadIdx.x & 31;
  const int tm = blockIdx.x * 8 + wave;
  if (tm >= N_NODES / 16) return;               // wave-uniform branch
  const int m0 = tm << 4;
  v8f c0 = {}, c1 = {};
#pragma unroll
  for (int ko = 0; ko < H2_DIM; ko += 32) {
    v16bf a = load_a_frag(h2, H2_DIM, m0, ko, lane);
    v16bf b0 = load_bt_frag(Wat, H2_DIM, 0, ko, lane);
    v16bf b1 = load_bt_frag(Wat, H2_DIM, 16, ko, lane);
    c0 = WMMA_BF16(a, b0, c0);
    c1 = WMMA_BF16(a, b1, c1);
  }
  const int col = lane & 15;
  const float ctx0 = ctx[col], ctx1 = ctx[col + 16];
  const float ba0 = ba[col], ba1 = ba[col + 16];
#pragma unroll
  for (int r = 0; r < 8; ++r) {
    float t = tanhf(c0[r] + ba0) * ctx0 + tanhf(c1[r] + ba1) * ctx1;
    // reduce the 16 lanes of this half-wave (wave32): rows 0-7 in lanes 0-15,
    // rows 8-15 in lanes 16-31
#pragma unroll
    for (int off = 8; off; off >>= 1) t += __shfl_xor(t, off, 16);
    if ((lane & 15) == 0) {
      int row = m0 + ((lane >> 4) << 3) + r;
      w_raw[row] = t + 0.4f * x[(size_t)row * D_IN + (D_IN - 1)];
    }
  }
}

// ---------------- softmax + weighted pooling --------------------------------
__global__ void max_kernel(const float* __restrict__ w, unsigned* __restrict__ key,
                           int n) {
  __shared__ float s[256];
  float m = -INFINITY;
  for (int i = blockIdx.x * blockDim.x + threadIdx.x; i < n;
       i += gridDim.x * blockDim.x)
    m = fmaxf(m, w[i]);
  s[threadIdx.x] = m;
  __syncthreads();
  for (int off = 128; off; off >>= 1) {
    if ((int)threadIdx.x < off)
      s[threadIdx.x] = fmaxf(s[threadIdx.x], s[threadIdx.x + off]);
    __syncthreads();
  }
  if (threadIdx.x == 0) {
    unsigned b = __float_as_uint(s[0]);
    unsigned k = (b & 0x80000000u) ? ~b : (b | 0x80000000u);
    atomicMax(key, k);
  }
}

__global__ void exp_sum_kernel(float* __restrict__ w,
                               const unsigned* __restrict__ key,
                               float* __restrict__ S, int n) {
  unsigned k = *key;
  unsigned b = (k & 0x80000000u) ? (k & 0x7FFFFFFFu) : ~k;
  const float mx = __uint_as_float(b);
  __shared__ float s[256];
  float part = 0.0f;
  for (int i = blockIdx.x * blockDim.x + threadIdx.x; i < n;
       i += gridDim.x * blockDim.x) {
    float e = __expf(w[i] - mx);
    w[i] = e;                                    // in-place: w now holds exp
    part += e;
  }
  s[threadIdx.x] = part;
  __syncthreads();
  for (int off = 128; off; off >>= 1) {
    if ((int)threadIdx.x < off) s[threadIdx.x] += s[threadIdx.x + off];
    __syncthreads();
  }
  if (threadIdx.x == 0) atomAddF(S, s[0]);
}

// P[f] = sum_n exp[n] * h2f[n,f]  (thread = feature lane, register accumulate)
__global__ void pool_kernel(const float* __restrict__ e,
                            const float* __restrict__ h2f,
                            float* __restrict__ P, int n) {
  const int f = threadIdx.x & (H2_DIM - 1);
  const int g = blockIdx.x * (blockDim.x / H2_DIM) + (threadIdx.x / H2_DIM);
  const int ng = gridDim.x * (blockDim.x / H2_DIM);
  float acc = 0.0f;
  for (int node = g; node < n; node += ng)
    acc += e[node] * h2f[(size_t)node * H2_DIM + f];
  __shared__ float sP[H2_DIM];
  if (threadIdx.x < H2_DIM) sP[threadIdx.x] = 0.0f;
  __syncthreads();
  atomAddLds(&sP[f], acc);
  __syncthreads();
  if (threadIdx.x < H2_DIM) atomAddF(&P[threadIdx.x], sP[threadIdx.x]);
}

// pooled = P/(S*N); z = relu(pooled@Wc1+bc1); out = sigmoid(z@Wc2+bc2)
__global__ void final_kernel(const float* __restrict__ S,
                             const float* __restrict__ P,
                             const float* __restrict__ Wc1,
                             const float* __restrict__ bc1,
                             const float* __restrict__ Wc2,
                             const float* __restrict__ bc2,
                             float* __restrict__ out) {
  __shared__ float pooled[H2_DIM];
  __shared__ float z[H2_DIM / 2];
  const int t = threadIdx.x;
  if (t < H2_DIM) pooled[t] = P[t] / ((*S) * (float)N_NODES);
  __syncthreads();
  if (t < H2_DIM / 2) {
    float acc = bc1[t];
#pragma unroll
    for (int f = 0; f < H2_DIM; ++f) acc += pooled[f] * Wc1[f * (H2_DIM / 2) + t];
    z[t] = fmaxf(acc, 0.0f);
  }
  __syncthreads();
  if (t == 0) {
    float acc = bc2[0];
#pragma unroll
    for (int j = 0; j < H2_DIM / 2; ++j) acc += z[j] * Wc2[j];
    out[0] = 1.0f / (1.0f + __expf(-acc));
  }
}

// ---------------- host launcher ----------------
extern "C" void kernel_launch(void* const* d_in, const int* in_sizes, int n_in,
                              void* d_out, int out_size, void* d_ws,
                              size_t ws_size, hipStream_t stream) {
  (void)in_sizes; (void)n_in; (void)out_size; (void)ws_size;
  const float* x    = (const float*)d_in[0];
  const int*   ei   = (const int*)d_in[1];
  const int*   src  = ei;
  const int*   dstv = ei + N_EDGES;
  const float* W1_l = (const float*)d_in[2];
  const float* W1_r = (const float*)d_in[3];
  const float* b1   = (const float*)d_in[4];
  const float* W2_l = (const float*)d_in[5];
  const float* W2_r = (const float*)d_in[6];
  const float* b2   = (const float*)d_in[7];
  const float* Wa   = (const float*)d_in[8];
  const float* ba   = (const float*)d_in[9];
  const float* ctx  = (const float*)d_in[10];
  const float* Wc1  = (const float*)d_in[11];
  const float* bc1  = (const float*)d_in[12];
  const float* Wc2  = (const float*)d_in[13];
  const float* bc2  = (const float*)d_in[14];
  float* out = (float*)d_out;

  // ---- workspace layout (lifetime-aliased) ----
  char* ws = (char*)d_ws;
  size_t off = 0;
  auto take = [&](size_t bytes) {
    size_t p = off;
    off += (bytes + 255) & ~(size_t)255;
    return p;
  };
  float*    deg    = (float*)(ws + take((size_t)N_NODES * 4));
  float*    w_raw  = (float*)(ws + take((size_t)N_NODES * 4));
  char*     scal   = ws + take(512);
  unsigned* maxkey = (unsigned*)scal;
  float*    Ssum   = (float*)(scal + 4);
  float*    Psum   = (float*)(scal + 8);            // 64 floats
  __bf16*   wbf    = (__bf16*)(ws + take((size_t)34816 * 2));
  __bf16*   W1l_t  = wbf;                 // [128 x 64]
  __bf16*   W1r_t  = wbf + 8192;          // [128 x 64]
  __bf16*   W2l_t  = wbf + 16384;         // [64 x 128]
  __bf16*   W2r_t  = wbf + 24576;         // [64 x 128]
  __bf16*   Wa_t   = wbf + 32768;         // [32 x 64]
  // region A: xbf + agg1bf (phase 1) -> agg2bf (phase 2)
  char*   regA   = ws + take((size_t)N_NODES * H1_DIM * 2);
  __bf16* xbf    = (__bf16*)regA;
  __bf16* agg1bf = (__bf16*)regA + (size_t)N_NODES * D_IN;
  __bf16* agg2bf = (__bf16*)regA;
  // region B: sum1 (phase 1) -> h2 f32 (phase 2)
  char*  regB = ws + take((size_t)N_NODES * D_IN * 4);
  float* sum1 = (float*)regB;
  float* h2f  = (float*)regB;
  __bf16* h1bf = (__bf16*)(ws + take((size_t)N_NODES * H1_DIM * 2));
  float*  sum2 = (float*)(ws + take((size_t)N_NODES * H1_DIM * 4));
  __bf16* h2bf = (__bf16*)(ws + take((size_t)N_NODES * H2_DIM * 2));

  // ---- zero init (graph-capturable async memsets) ----
  hipMemsetAsync(deg, 0, (size_t)N_NODES * 4, stream);
  hipMemsetAsync(scal, 0, 512, stream);
  hipMemsetAsync(sum1, 0, (size_t)N_NODES * D_IN * 4, stream);
  hipMemsetAsync(sum2, 0, (size_t)N_NODES * H1_DIM * 4, stream);

  const int B = 256;
  // bf16 conversions (weights transposed to [Nout x K])
  cvt_bf16_kernel<<<(N_NODES * D_IN + B - 1) / B, B, 0, stream>>>(x, xbf,
                                                                  N_NODES * D_IN);
  cvt_transpose_kernel<<<32, B, 0, stream>>>(W1_l, W1l_t, 6, H1_DIM,
                                             D_IN * H1_DIM);
  cvt_transpose_kernel<<<32, B, 0, stream>>>(W1_r, W1r_t, 6, H1_DIM,
                                             D_IN * H1_DIM);
  cvt_transpose_kernel<<<32, B, 0, stream>>>(W2_l, W2l_t, 7, H2_DIM,
                                             H1_DIM * H2_DIM);
  cvt_transpose_kernel<<<32, B, 0, stream>>>(W2_r, W2r_t, 7, H2_DIM,
                                             H1_DIM * H2_DIM);
  cvt_transpose_kernel<<<8, B, 0, stream>>>(Wa, Wa_t, 6, ATT_DIM,
                                            H2_DIM * ATT_DIM);

  // degree
  deg_kernel<<<(N_EDGES + B - 1) / B, B, 0, stream>>>(dstv, deg, N_EDGES);

  // layer 1 aggregation + GEMM
  {
    long long tot = (long long)N_EDGES * (D_IN / 4);
    scatter_add_kernel<float><<<(unsigned)((tot + B - 1) / B), B, 0, stream>>>(
        x, src, dstv, sum1, N_EDGES, 6);
    long long ntot = (long long)N_NODES * D_IN;
    finalize_kernel<<<(unsigned)((ntot + B - 1) / B), B, 0, stream>>>(
        sum1, deg, agg1bf, ntot, 6);
  }
  gemm1_kernel<<<N_NODES / 16, B, 0, stream>>>(agg1bf, xbf, W1l_t, W1r_t, b1,
                                               h1bf);

  // layer 2 aggregation + GEMM
  {
    long long tot = (long long)N_EDGES * (H1_DIM / 4);
    scatter_add_kernel<__bf16><<<(unsigned)((tot + B - 1) / B), B, 0, stream>>>(
        h1bf, src, dstv, sum2, N_EDGES, 7);
    long long ntot = (long long)N_NODES * H1_DIM;
    finalize_kernel<<<(unsigned)((ntot + B - 1) / B), B, 0, stream>>>(
        sum2, deg, agg2bf, ntot, 7);
  }
  gemm2_kernel<<<(N_NODES / 16) * (H2_DIM / 16) / 8, B, 0, stream>>>(
      agg2bf, h1bf, W2l_t, W2r_t, b2, h2f, h2bf);

  // attention scores -> raw weights
  gemm3_kernel<<<(N_NODES / 16 + 7) / 8, B, 0, stream>>>(h2bf, Wa_t, ba, ctx, x,
                                                         w_raw);

  // softmax + weighted pool
  max_kernel<<<256, B, 0, stream>>>(w_raw, maxkey, N_NODES);
  exp_sum_kernel<<<256, B, 0, stream>>>(w_raw, maxkey, Ssum, N_NODES);
  pool_kernel<<<256, B, 0, stream>>>(w_raw, h2f, Psum, N_NODES);

  // classifier head
  final_kernel<<<1, 64, 0, stream>>>(Ssum, Psum, Wc1, bc1, Wc2, bc2, out);
}
